// S4Enc_84997402787917
// MI455X (gfx1250) — matrix operand order repaired and checked by reading, not compiled, not run on hardware
//
#include <hip/hip_runtime.h>
#include <stdint.h>

// ---------------- problem constants ----------------
#define BB   16
#define LL   2048
#define FIN  75
#define HH   256
#define NN   64
#define NL   4
#define REPR 512

typedef __attribute__((ext_vector_type(16))) __bf16 v16bf;
typedef __attribute__((ext_vector_type(8)))  float  v8f;

__device__ __forceinline__ uint16_t f2bf(float f) {
    uint32_t u = __float_as_uint(f);
    uint32_t r = u + 0x7FFFu + ((u >> 16) & 1u);
    return (uint16_t)(r >> 16);
}

union frag_u  { uint4 q[2]; v16bf v; };
union pack8_u { uint16_t s[8]; uint4 q; };

// ---------------- kernel 0: convert weights to bf16 ----------------
// out_W (NL,512,256) -> same-layout bf16; dec_W (256,512) -> transposed bf16 (512,256)
__global__ void cvt_weights(const float* __restrict__ outW,
                            const float* __restrict__ decW,
                            uint16_t* __restrict__ wbf,
                            uint16_t* __restrict__ decWT) {
    int idx = blockIdx.x * blockDim.x + threadIdx.x;
    const int n1 = NL * 2 * HH * HH;        // 524288
    if (idx < n1) wbf[idx] = f2bf(outW[idx]);
    if (idx < HH * REPR) {                  // 131072
        int j = idx / HH, k = idx % HH;     // decWT[j][k] = dec_W[k][j]
        decWT[j * HH + k] = f2bf(decW[k * REPR + j]);
    }
}

// ---------------- kernel 1: embed ----------------
// h[b][c][l] = x[b][l][:75] . embed_W[:,c] + embed_b[c]
__global__ void embed_kernel(const float* __restrict__ x,
                             const float* __restrict__ eW,
                             const float* __restrict__ eb,
                             float* __restrict__ h) {
    int l  = blockIdx.x * blockDim.x + threadIdx.x;
    int c  = blockIdx.y;
    int b  = blockIdx.z;
    const float* xr = x + ((long)b * LL + l) * FIN;
    float acc = eb[c];
    #pragma unroll 5
    for (int k = 0; k < FIN; ++k) acc += xr[k] * eW[k * HH + c];
    h[((long)b * HH + c) * LL + l] = acc;
}

// ---------------- kernel 2: prenorm (over H, h layout (B,H,L)) ----------------
__global__ void ln_kernel(const float* __restrict__ h,
                          const float* __restrict__ g,
                          const float* __restrict__ bb,
                          float* __restrict__ z) {
    int t = blockIdx.x * blockDim.x + threadIdx.x;   // over B*L
    int b = t / LL, l = t % LL;
    const float* base = h + (long)b * HH * LL + l;
    float s = 0.f, s2 = 0.f;
    for (int c = 0; c < HH; ++c) { float v = base[c * LL]; s += v; s2 += v * v; }
    float m = s * (1.f / HH);
    float var = s2 * (1.f / HH) - m * m;
    float rstd = rsqrtf(var + 1e-5f);
    float* zb = z + (long)b * HH * LL + l;
    for (int c = 0; c < HH; ++c)
        zb[c * LL] = (base[c * LL] - m) * rstd * g[c] + bb[c];
}

// ---------------- kernel 3: S4D recurrence scan + D-skip + GELU -> bf16 (transposed out) ----
// one wave32 per (b,c); 2 complex states per lane; butterfly reduce per step.
// Output layout: ybfT[(b*L + l)*H + c]  (so GEMM B-fragments are K-contiguous).
__global__ void s4_scan(const float* __restrict__ z,
                        const float* __restrict__ log_dt,
                        const float* __restrict__ Alre,
                        const float* __restrict__ Aim,
                        const float* __restrict__ Cre,
                        const float* __restrict__ Cim,
                        const float* __restrict__ Dsk,
                        uint16_t* __restrict__ ybfT) {
    int wg   = blockIdx.x * 4 + (threadIdx.x >> 5);
    int lane = threadIdx.x & 31;
    int b = wg / HH, c = wg % HH;

    float dt = expf(log_dt[c]);
    float w_r[2], w_i[2], cd_r[2], cd_i[2], s_r[2] = {0.f, 0.f}, s_i[2] = {0.f, 0.f};
    #pragma unroll
    for (int j = 0; j < 2; ++j) {
        int n = lane * 2 + j;
        float ar = -expf(Alre[c * NN + n]);
        float ai = Aim[c * NN + n];
        float er = expf(ar * dt);
        float wr = er * cosf(ai * dt);
        float wi = er * sinf(ai * dt);
        w_r[j] = wr; w_i[j] = wi;
        float cr = Cre[c * NN + n], ci = Cim[c * NN + n];
        float nr = cr * (wr - 1.f) - ci * wi;        // C*(w-1)
        float ni = cr * wi + ci * (wr - 1.f);
        float den = 1.f / (ar * ar + ai * ai);
        cd_r[j] = (nr * ar + ni * ai) * den;          // num*conj(A)/|A|^2
        cd_i[j] = (ni * ar - nr * ai) * den;
    }
    float D = Dsk[c];
    const float* zr = z + ((long)b * HH + c) * LL;
    uint16_t*   yr = ybfT + (long)b * LL * HH + c;

    for (int l0 = 0; l0 < LL; l0 += 32) {
        float uvec = zr[l0 + lane];                   // coalesced 32 timesteps
        float ysave = 0.f;
        for (int j = 0; j < 32; ++j) {
            float u = __shfl(uvec, j, 32);
            #pragma unroll
            for (int k = 0; k < 2; ++k) {
                float tr = w_r[k] * s_r[k] - w_i[k] * s_i[k] + u;
                float ti = w_r[k] * s_i[k] + w_i[k] * s_r[k];
                s_r[k] = tr; s_i[k] = ti;
            }
            float p = cd_r[0] * s_r[0] - cd_i[0] * s_i[0]
                    + cd_r[1] * s_r[1] - cd_i[1] * s_i[1];
            #pragma unroll
            for (int m = 16; m >= 1; m >>= 1) p += __shfl_xor(p, m, 32);
            float y = 2.f * p + D * u;
            // tanh-GELU (matches jax.nn.gelu default)
            y = 0.5f * y * (1.f + tanhf(0.7978845608f * (y + 0.044715f * y * y * y)));
            if (lane == j) ysave = y;
        }
        yr[(long)(l0 + lane) * HH] = f2bf(ysave);     // transposed store (L2-resident)
    }
}

// ---------------- kernel 4: fused GLU out-projection + residual (WMMA bf16, no LDS) ------
// per b: zz = W(512x256) @ y_b(256xL); h += zz[g]*sigmoid(zz[g+256])
// On the last layer, also emit the residual result as bf16 transposed (for decode GEMM).
__global__ void glu_gemm(const uint16_t* __restrict__ Wbf,   // layer's (512,256) bf16
                         const float* __restrict__ outb,      // (512)
                         const uint16_t* __restrict__ ybfT,   // (B,L,H) bf16
                         float* __restrict__ h,               // (B,H,L) fp32, in/out
                         uint16_t* __restrict__ hbfT,         // (B,L,H) bf16 out
                         int emit_t) {
    int b  = blockIdx.z;
    int l0 = blockIdx.x * 16;
    int wave = threadIdx.x >> 5, lane = threadIdx.x & 31;
    int half = lane >> 4, nn = lane & 15;
    int g0 = blockIdx.y * 128 + wave * 16;

    const uint16_t* brow0 = ybfT + ((long)b * LL + l0 + nn) * HH + half * 8;
    const uint16_t* a1row = Wbf + (g0 + nn) * HH;
    const uint16_t* a2row = Wbf + (g0 + 256 + nn) * HH;

    v8f acc1 = {}; v8f acc2 = {};
    #pragma unroll
    for (int k0 = 0; k0 < HH; k0 += 32) {
        frag_u bf, a1, a2;
        // B fragment: two contiguous 8-value K runs at k0+8*half and k0+16+8*half for column nn
        bf.q[0] = *(const uint4*)(brow0 + k0);
        bf.q[1] = *(const uint4*)(brow0 + k0 + 16);
        // A fragments: two contiguous 8-value K runs at k0+8*half and k0+16+8*half
        a1.q[0] = *(const uint4*)(a1row + k0 + half * 8);
        a1.q[1] = *(const uint4*)(a1row + k0 + 16 + half * 8);
        a2.q[0] = *(const uint4*)(a2row + k0 + half * 8);
        a2.q[1] = *(const uint4*)(a2row + k0 + 16 + half * 8);
        acc1 = __builtin_amdgcn_wmma_f32_16x16x32_bf16(false, a1.v, false, bf.v,
                                                       (short)0, acc1, false, false);
        acc2 = __builtin_amdgcn_wmma_f32_16x16x32_bf16(false, a2.v, false, bf.v,
                                                       (short)0, acc2, false, false);
    }
    // epilogue: GLU + residual. acc[r] -> row g0 + r + 8*half, col l0+nn
    pack8_u pk;
    #pragma unroll
    for (int r = 0; r < 8; ++r) {
        int gi = g0 + r + half * 8;
        float z1 = acc1[r] + outb[gi];
        float z2 = acc2[r] + outb[gi + 256];
        float v = z1 / (1.f + expf(-z2));
        float* hp = h + ((long)b * HH + gi) * LL + l0 + nn;
        float nh = *hp + v;
        *hp = nh;
        pk.s[r] = f2bf(nh);
    }
    if (emit_t) {
        // rows g0+8*half .. +7 are contiguous in c: one 16B transposed store
        *(uint4*)(hbfT + ((long)b * LL + l0 + nn) * HH + g0 + half * 8) = pk.q;
    }
}

// ---------------- kernel 5: decode GEMM (WMMA bf16, no LDS) ----------------
// out[b,l,j] = sum_c h[b,c,l]*dec_W[c,j] + dec_b[j], via A=dec_W^T, B=hbfT
__global__ void dec_gemm(const uint16_t* __restrict__ WT,    // (512,256) bf16 (dec_W^T)
                         const float* __restrict__ decb,     // (512)
                         const uint16_t* __restrict__ hbfT,  // (B,L,H) bf16
                         float* __restrict__ out) {          // (B,L,512)
    int b  = blockIdx.z;
    int l0 = blockIdx.x * 16;
    int wave = threadIdx.x >> 5, lane = threadIdx.x & 31;
    int half = lane >> 4, nn = lane & 15;
    int j0 = blockIdx.y * 128 + wave * 16;

    const uint16_t* brow0 = hbfT + ((long)b * LL + l0 + nn) * HH + half * 8;
    const uint16_t* arow  = WT + (j0 + nn) * HH;

    v8f acc = {};
    #pragma unroll
    for (int k0 = 0; k0 < HH; k0 += 32) {
        frag_u bf, a;
        bf.q[0] = *(const uint4*)(brow0 + k0);
        bf.q[1] = *(const uint4*)(brow0 + k0 + 16);
        a.q[0]  = *(const uint4*)(arow + k0 + half * 8);
        a.q[1]  = *(const uint4*)(arow + k0 + 16 + half * 8);
        acc = __builtin_amdgcn_wmma_f32_16x16x32_bf16(false, a.v, false, bf.v,
                                                      (short)0, acc, false, false);
    }
    #pragma unroll
    for (int r = 0; r < 8; ++r) {
        int ji = j0 + r + half * 8;
        out[((long)b * LL + l0 + nn) * REPR + ji] = acc[r] + decb[ji];
    }
}

// ---------------- host orchestration ----------------
extern "C" void kernel_launch(void* const* d_in, const int* in_sizes, int n_in,
                              void* d_out, int out_size, void* d_ws, size_t ws_size,
                              hipStream_t stream) {
    const float* x      = (const float*)d_in[0];
    const float* eW     = (const float*)d_in[1];
    const float* eb     = (const float*)d_in[2];
    const float* log_dt = (const float*)d_in[3];
    const float* Alre   = (const float*)d_in[4];
    const float* Aim    = (const float*)d_in[5];
    const float* Cre    = (const float*)d_in[6];
    const float* Cim    = (const float*)d_in[7];
    const float* Dsk    = (const float*)d_in[8];
    const float* outW   = (const float*)d_in[9];
    const float* outb   = (const float*)d_in[10];
    const float* lng    = (const float*)d_in[11];
    const float* lnb    = (const float*)d_in[12];
    const float* decW   = (const float*)d_in[13];
    const float* decb   = (const float*)d_in[14];
    float* out = (float*)d_out;

    char* ws = (char*)d_ws;
    const size_t SZ_H  = (size_t)BB * HH * LL * sizeof(float);   // 32 MB
    const size_t SZ_BF = (size_t)BB * HH * LL * 2;               // 16 MB
    float*    h     = (float*)(ws);
    float*    z     = (float*)(ws + SZ_H);
    uint16_t* ybfT  = (uint16_t*)(ws + 2 * SZ_H);
    uint16_t* hbfT  = (uint16_t*)(ws + 2 * SZ_H + SZ_BF);
    uint16_t* wbf   = (uint16_t*)(ws + 2 * SZ_H + 2 * SZ_BF);
    uint16_t* decWT = wbf + (size_t)NL * 2 * HH * HH;

    // weights -> bf16 (and dec_W transpose)
    cvt_weights<<<dim3((NL * 2 * HH * HH) / 256), 256, 0, stream>>>(outW, decW, wbf, decWT);
    // embed
    embed_kernel<<<dim3(LL / 256, HH, BB), 256, 0, stream>>>(x, eW, eb, h);

    for (int i = 0; i < NL; ++i) {
        ln_kernel<<<dim3((BB * LL) / 256), 256, 0, stream>>>(h, lng + i * HH, lnb + i * HH, z);
        s4_scan<<<dim3((BB * HH) / 4), 128, 0, stream>>>(
            z, log_dt + i * HH, Alre + (size_t)i * HH * NN, Aim + (size_t)i * HH * NN,
            Cre + (size_t)i * HH * NN, Cim + (size_t)i * HH * NN, Dsk + i * HH, ybfT);
        glu_gemm<<<dim3(LL / 16, 2, BB), 256, 0, stream>>>(
            wbf + (size_t)i * 2 * HH * HH, outb + i * 2 * HH, ybfT, h,
            hbfT, (i == NL - 1) ? 1 : 0);
    }
    dec_gemm<<<dim3(LL / 16, REPR / 128, BB), 256, 0, stream>>>(decWT, decb, hbfT, out);
}
